// MeshProcessorBlock_4552665334037
// MI455X (gfx1250) — matrix operand order repaired and checked by reading, not compiled, not run on hardware
//
#include <hip/hip_runtime.h>
#include <hip/hip_bf16.h>
#include <cstdint>

typedef __attribute__((ext_vector_type(16))) __bf16   v16bf;
typedef __attribute__((ext_vector_type(8)))  float    v8f;
typedef __attribute__((ext_vector_type(4)))  uint32_t u32x4;

#define NN_NODES 50000
#define NN_EDGES 400000
#define DD 128
#define TILE 32
#define EPSL 1e-5f

struct Frag16 { union { v16bf bf; u32x4 q[2]; }; };

__device__ __forceinline__ uint32_t f2bf1(float f) {
  uint32_t u = __float_as_uint(f);
  return (u + 0x7FFFu + ((u >> 16) & 1u)) >> 16;   // round-to-nearest-even bf16
}
__device__ __forceinline__ uint32_t pack_bf2(float lo, float hi) {
  return f2bf1(lo) | (f2bf1(hi) << 16);
}
__device__ __forceinline__ float silu_f(float x) {
  return x * (1.0f / (1.0f + __expf(-x)));
}

// One wave computes a 16-wide N column (n = wave*16..+15) of a [32 x 128] output:
// two 16x16 M tiles sharing the same B fragments. A comes from LDS (bf16 rows),
// B comes from pre-packed global weights in exact WMMA fragment order.
template<int KSTEPS>
__device__ __forceinline__ void mma_rows(const unsigned char* Xs, int ldxB,
                                         const u32x4* __restrict__ wpack,
                                         int wave, int lane, v8f& acc0, v8f& acc1) {
  const int hf   = lane >> 4;     // half-wave selects K sub-range
  const int mrow = lane & 15;
#pragma unroll
  for (int ks = 0; ks < KSTEPS; ++ks) {
    Frag16 B;
    const u32x4* bp = wpack + (size_t)((wave * KSTEPS + ks) * 32 + lane) * 2;
    B.q[0] = bp[0];
    B.q[1] = bp[1];
    // A layout (ISA 7.12.2, 16-bit A 16x32): lane<16 -> K {0..7,16..23}, lane>=16 -> +8
    const int kbyte = (ks * 32 + 8 * hf) * 2;
    Frag16 A0, A1;
    const unsigned char* a0 = Xs + mrow * ldxB + kbyte;
    const unsigned char* a1 = Xs + (16 + mrow) * ldxB + kbyte;
    A0.q[0] = *(const u32x4*)(a0);
    A0.q[1] = *(const u32x4*)(a0 + 32);
    A1.q[0] = *(const u32x4*)(a1);
    A1.q[1] = *(const u32x4*)(a1 + 32);
    acc0 = __builtin_amdgcn_wmma_f32_16x16x32_bf16(false, A0.bf, false, B.bf,
                                                   (short)0, acc0, false, false);
    acc1 = __builtin_amdgcn_wmma_f32_16x16x32_bf16(false, A1.bf, false, B.bf,
                                                   (short)0, acc1, false, false);
  }
}

__global__ void zero_kernel(float4* p, int n4) {
  int i = blockIdx.x * 256 + threadIdx.x;
  if (i < n4) p[i] = make_float4(0.f, 0.f, 0.f, 0.f);
}

// Repack W[K x 128] (f32, row-major) into bf16 WMMA B-fragment order:
// u32 index ((tn*KS+ks)*32+lane)*8+v holds {W[k][n], W[k+1][n]} with
// k = ks*32 + 2v + 8*(lane>>4) + 8*(v>>2), n = tn*16 + (lane&15).
__global__ void pack_kernel(const float* __restrict__ W, uint32_t* __restrict__ out, int K) {
  const int KS = K / 32;
  int t = blockIdx.x * 256 + threadIdx.x;
  if (t >= 8 * KS * 32 * 8) return;
  int v    = t & 7;
  int lane = (t >> 3) & 31;
  int rest = t >> 8;
  int ks = rest % KS;
  int tn = rest / KS;
  int k = ks * 32 + 2 * v + 8 * (lane >> 4) + 8 * (v >> 2);
  int n = tn * 16 + (lane & 15);
  out[t] = pack_bf2(W[(size_t)k * DD + n], W[(size_t)(k + 1) * DD + n]);
}

__global__ __launch_bounds__(256)
void edge_kernel(const float* __restrict__ efeat, const float* __restrict__ nfeat,
                 const int* __restrict__ src, const int* __restrict__ dst,
                 const u32x4* __restrict__ w1p, const u32x4* __restrict__ w2p,
                 const float* __restrict__ b1, const float* __restrict__ b2,
                 const float* __restrict__ gw, const float* __restrict__ bnw,
                 float* __restrict__ out_e, float* __restrict__ agg) {
  __shared__ __align__(16) unsigned char smem[TILE * 384 * 2 + TILE * DD * 2 + TILE * 8];
  unsigned char* Xs = smem;                                           // 24576 B (bf16 X)
  unsigned short* Hs = (unsigned short*)(smem + TILE * 384 * 2);      // 8192 B (bf16 H)
  float* stats = (float*)(smem + TILE * 384 * 2 + TILE * DD * 2);     // 64 f32
  float* Ys = (float*)smem;                                           // 16384 B, aliases Xs

  const int tid = threadIdx.x;
  const int e0 = blockIdx.x * TILE;

  // ---- stage X = [efeat | nfeat[src] | nfeat[dst]] as bf16 ----
  for (int i = tid; i < TILE * 96; i += 256) {     // 96 float4 per 384-wide row
    int row = i / 96, c4 = i % 96;
    int e = e0 + row;                              // E % TILE == 0, no guard needed
    const float4* p;
    if (c4 < 32)      p = (const float4*)(efeat + (size_t)e * DD) + c4;
    else if (c4 < 64) p = (const float4*)(nfeat + (size_t)src[e] * DD) + (c4 - 32);
    else              p = (const float4*)(nfeat + (size_t)dst[e] * DD) + (c4 - 64);
    float4 v = *p;
    uint32_t* q = (uint32_t*)(Xs + row * 768) + c4 * 2;
    q[0] = pack_bf2(v.x, v.y);
    q[1] = pack_bf2(v.z, v.w);
  }
  __syncthreads();

  const int wave = tid >> 5, lane = tid & 31;
  const int hf = lane >> 4;
  const int n = wave * 16 + (lane & 15);

  // ---- GEMM1: [32x384] @ [384x128] -> SiLU -> Hs ----
  v8f acc0 = {0,0,0,0,0,0,0,0}, acc1 = {0,0,0,0,0,0,0,0};
  mma_rows<12>(Xs, 768, w1p, wave, lane, acc0, acc1);
  {
    float bias = b1[n];
#pragma unroll
    for (int r = 0; r < 8; ++r) {
      int m0 = r + 8 * hf;
      Hs[m0 * DD + n]        = (unsigned short)f2bf1(silu_f(acc0[r] + bias));
      Hs[(16 + m0) * DD + n] = (unsigned short)f2bf1(silu_f(acc1[r] + bias));
    }
  }
  __syncthreads();

  // ---- GEMM2: [32x128] @ [128x128] -> Ys (f32, aliased over Xs) ----
  acc0 = (v8f){0,0,0,0,0,0,0,0};
  acc1 = (v8f){0,0,0,0,0,0,0,0};
  mma_rows<4>((const unsigned char*)Hs, 256, w2p, wave, lane, acc0, acc1);
  {
    float bias = b2[n];
#pragma unroll
    for (int r = 0; r < 8; ++r) {
      int m0 = r + 8 * hf;
      Ys[m0 * DD + n]        = acc0[r] + bias;
      Ys[(16 + m0) * DD + n] = acc1[r] + bias;
    }
  }
  __syncthreads();

  // ---- LayerNorm stats: 8 threads per row, combine with wave shuffles ----
  {
    int row = tid >> 3, sub = tid & 7;
    float s = 0.f, s2 = 0.f;
    const float* yr = Ys + row * DD + sub * 16;
#pragma unroll
    for (int j = 0; j < 16; ++j) { float y = yr[j]; s += y; s2 += y * y; }
#pragma unroll
    for (int o = 1; o < 8; o <<= 1) {
      s  += __shfl_xor(s,  o, 8);
      s2 += __shfl_xor(s2, o, 8);
    }
    if (sub == 0) {
      float mu  = s * (1.0f / DD);
      float var = s2 * (1.0f / DD) - mu * mu;
      stats[row * 2]     = mu;
      stats[row * 2 + 1] = rsqrtf(var + EPSL);
    }
  }
  __syncthreads();

  // ---- normalize + residual + store + scatter-sum (f32 atomics into agg) ----
  for (int i = tid; i < TILE * 32; i += 256) {     // float4 granularity
    int row = i >> 5, c4 = i & 31;
    int e = e0 + row;
    float mu = stats[row * 2], rs = stats[row * 2 + 1];
    float4 y = *((const float4*)(Ys + row * DD) + c4);
    float4 rv = *((const float4*)(efeat + (size_t)e * DD) + c4);
    int c = c4 * 4;
    float4 o;
    o.x = (y.x - mu) * rs * gw[c]     + bnw[c]     + rv.x;
    o.y = (y.y - mu) * rs * gw[c + 1] + bnw[c + 1] + rv.y;
    o.z = (y.z - mu) * rs * gw[c + 2] + bnw[c + 2] + rv.z;
    o.w = (y.w - mu) * rs * gw[c + 3] + bnw[c + 3] + rv.w;
    *((float4*)(out_e + (size_t)e * DD) + c4) = o;
    float* ap = agg + (size_t)dst[e] * DD + c;
    atomicAdd(ap + 0, o.x);
    atomicAdd(ap + 1, o.y);
    atomicAdd(ap + 2, o.z);
    atomicAdd(ap + 3, o.w);
  }
}

__global__ __launch_bounds__(256)
void node_kernel(const float* __restrict__ nfeat, const float* __restrict__ agg,
                 const u32x4* __restrict__ w1p, const u32x4* __restrict__ w2p,
                 const float* __restrict__ b1, const float* __restrict__ b2,
                 const float* __restrict__ gw, const float* __restrict__ bnw,
                 float* __restrict__ out_n) {
  __shared__ __align__(16) unsigned char smem[TILE * 256 * 2 + TILE * DD * 2 + TILE * 8];
  unsigned char* Xs = smem;                                           // 16384 B
  unsigned short* Hs = (unsigned short*)(smem + TILE * 256 * 2);      // 8192 B
  float* stats = (float*)(smem + TILE * 256 * 2 + TILE * DD * 2);
  float* Ys = (float*)smem;                                           // aliases Xs

  const int tid = threadIdx.x;
  const int v0 = blockIdx.x * TILE;

  // ---- stage X = [agg | nfeat] as bf16 (zero-pad tail rows) ----
  for (int i = tid; i < TILE * 64; i += 256) {     // 64 float4 per 256-wide row
    int row = i >> 6, c4 = i & 63;
    int nd = v0 + row;
    float4 v = make_float4(0.f, 0.f, 0.f, 0.f);
    if (nd < NN_NODES) {
      const float4* p = (c4 < 32)
          ? (const float4*)(agg + (size_t)nd * DD) + c4
          : (const float4*)(nfeat + (size_t)nd * DD) + (c4 - 32);
      v = *p;
    }
    uint32_t* q = (uint32_t*)(Xs + row * 512) + c4 * 2;
    q[0] = pack_bf2(v.x, v.y);
    q[1] = pack_bf2(v.z, v.w);
  }
  __syncthreads();

  const int wave = tid >> 5, lane = tid & 31;
  const int hf = lane >> 4;
  const int n = wave * 16 + (lane & 15);

  v8f acc0 = {0,0,0,0,0,0,0,0}, acc1 = {0,0,0,0,0,0,0,0};
  mma_rows<8>(Xs, 512, w1p, wave, lane, acc0, acc1);
  {
    float bias = b1[n];
#pragma unroll
    for (int r = 0; r < 8; ++r) {
      int m0 = r + 8 * hf;
      Hs[m0 * DD + n]        = (unsigned short)f2bf1(silu_f(acc0[r] + bias));
      Hs[(16 + m0) * DD + n] = (unsigned short)f2bf1(silu_f(acc1[r] + bias));
    }
  }
  __syncthreads();

  acc0 = (v8f){0,0,0,0,0,0,0,0};
  acc1 = (v8f){0,0,0,0,0,0,0,0};
  mma_rows<4>((const unsigned char*)Hs, 256, w2p, wave, lane, acc0, acc1);
  {
    float bias = b2[n];
#pragma unroll
    for (int r = 0; r < 8; ++r) {
      int m0 = r + 8 * hf;
      Ys[m0 * DD + n]        = acc0[r] + bias;
      Ys[(16 + m0) * DD + n] = acc1[r] + bias;
    }
  }
  __syncthreads();

  {
    int row = tid >> 3, sub = tid & 7;
    float s = 0.f, s2 = 0.f;
    const float* yr = Ys + row * DD + sub * 16;
#pragma unroll
    for (int j = 0; j < 16; ++j) { float y = yr[j]; s += y; s2 += y * y; }
#pragma unroll
    for (int o = 1; o < 8; o <<= 1) {
      s  += __shfl_xor(s,  o, 8);
      s2 += __shfl_xor(s2, o, 8);
    }
    if (sub == 0) {
      float mu  = s * (1.0f / DD);
      float var = s2 * (1.0f / DD) - mu * mu;
      stats[row * 2]     = mu;
      stats[row * 2 + 1] = rsqrtf(var + EPSL);
    }
  }
  __syncthreads();

  for (int i = tid; i < TILE * 32; i += 256) {
    int row = i >> 5, c4 = i & 31;
    int nd = v0 + row;
    if (nd >= NN_NODES) continue;
    float mu = stats[row * 2], rs = stats[row * 2 + 1];
    float4 y = *((const float4*)(Ys + row * DD) + c4);
    float4 rv = *((const float4*)(nfeat + (size_t)nd * DD) + c4);
    int c = c4 * 4;
    float4 o;
    o.x = (y.x - mu) * rs * gw[c]     + bnw[c]     + rv.x;
    o.y = (y.y - mu) * rs * gw[c + 1] + bnw[c + 1] + rv.y;
    o.z = (y.z - mu) * rs * gw[c + 2] + bnw[c + 2] + rv.z;
    o.w = (y.w - mu) * rs * gw[c + 3] + bnw[c + 3] + rv.w;
    *((float4*)(out_n + (size_t)nd * DD) + c4) = o;
  }
}

extern "C" void kernel_launch(void* const* d_in, const int* in_sizes, int n_in,
                              void* d_out, int out_size, void* d_ws, size_t ws_size,
                              hipStream_t stream) {
  const float* efeat = (const float*)d_in[0];
  const float* nfeat = (const float*)d_in[1];
  const int*   src   = (const int*)d_in[2];
  const int*   dst   = (const int*)d_in[3];
  // edge_mlp: W1 b1 W2 b2 g bn ; node_mlp: same order
  const float* eW1 = (const float*)d_in[4];
  const float* eb1 = (const float*)d_in[5];
  const float* eW2 = (const float*)d_in[6];
  const float* eb2 = (const float*)d_in[7];
  const float* eg  = (const float*)d_in[8];
  const float* ebn = (const float*)d_in[9];
  const float* nW1 = (const float*)d_in[10];
  const float* nb1 = (const float*)d_in[11];
  const float* nW2 = (const float*)d_in[12];
  const float* nb2 = (const float*)d_in[13];
  const float* ng  = (const float*)d_in[14];
  const float* nbn = (const float*)d_in[15];

  unsigned char* ws = (unsigned char*)d_ws;
  float* agg = (float*)ws;
  size_t off = (size_t)NN_NODES * DD * sizeof(float);       // 25.6 MB
  uint32_t* w1ep = (uint32_t*)(ws + off); off += (size_t)8 * 12 * 32 * 8 * 4;
  uint32_t* w2ep = (uint32_t*)(ws + off); off += (size_t)8 * 4  * 32 * 8 * 4;
  uint32_t* w1np = (uint32_t*)(ws + off); off += (size_t)8 * 8  * 32 * 8 * 4;
  uint32_t* w2np = (uint32_t*)(ws + off);

  float* out_e = (float*)d_out;
  float* out_n = (float*)d_out + (size_t)NN_EDGES * DD;

  // 1) zero scatter buffer (every call — workspace is not re-poisoned)
  int n4 = NN_NODES * DD / 4;
  zero_kernel<<<dim3((n4 + 255) / 256), dim3(256), 0, stream>>>((float4*)agg, n4);

  // 2) pack weights to bf16 WMMA fragment order
  pack_kernel<<<dim3((8 * 12 * 32 * 8 + 255) / 256), dim3(256), 0, stream>>>(eW1, w1ep, 384);
  pack_kernel<<<dim3((8 * 4  * 32 * 8 + 255) / 256), dim3(256), 0, stream>>>(eW2, w2ep, 128);
  pack_kernel<<<dim3((8 * 8  * 32 * 8 + 255) / 256), dim3(256), 0, stream>>>(nW1, w1np, 256);
  pack_kernel<<<dim3((8 * 4  * 32 * 8 + 255) / 256), dim3(256), 0, stream>>>(nW2, w2np, 128);

  // 3) edge update + scatter
  edge_kernel<<<dim3(NN_EDGES / TILE), dim3(256), 0, stream>>>(
      efeat, nfeat, src, dst, (const u32x4*)w1ep, (const u32x4*)w2ep,
      eb1, eb2, eg, ebn, out_e, agg);

  // 4) node update
  node_kernel<<<dim3((NN_NODES + TILE - 1) / TILE), dim3(256), 0, stream>>>(
      nfeat, agg, (const u32x4*)w1np, (const u32x4*)w2np,
      nb1, nb2, ng, nbn, out_n);
}
